// Residual_block_76038101008589
// MI455X (gfx1250) — matrix operand order repaired and checked by reading, not compiled, not run on hardware
//
#include <hip/hip_runtime.h>
#include <math.h>

typedef __attribute__((ext_vector_type(16))) _Float16 v16h;
typedef __attribute__((ext_vector_type(8)))  float    v8f;

#define HWC   65536      // H*W = 256*256
#define BATCH 4
#define NWIN  1024       // windows per image (32*32)

// ---- WMMA fragment index maps (per CDNA5 ISA VGPR layout tables) ----
// A (16x32 f16, MxK): lane<16 holds row m=lane, K = {0..7, 16..23};
//                     lane>=16 holds row m=lane-16, K = {8..15, 24..31}
__device__ __forceinline__ int amap(int lane, int i) {
  return (i < 8 ? i : i + 8) + ((lane & 16) ? 8 : 0);
}
// B (32x16 f16, KxN): lane<16 -> col n=lane, K = {0..7, 16..23}; lane>=16 -> n=lane-16, K = {8..15, 24..31}
__device__ __forceinline__ int bmapK(int lane, int i) {
  return (i < 8 ? i : i + 8) + ((lane & 16) ? 8 : 0);
}

__device__ __forceinline__ float lrelu_(float x) { return x >= 0.f ? x : 0.1f * x; }
__device__ __forceinline__ float gelu_(float x)  { return 0.5f * x * (1.f + erff(x * 0.70710678118654752f)); }
__device__ __forceinline__ float sigm_(float x)  { return 1.f / (1.f + expf(-x)); }

// =====================================================================
// Generic pointwise conv (NCHW): out[b,o,p] = act( sum_c W[o,c]*in[b,c,p] + bias[o] )
// One wave handles a 16-pixel tile for all COUT channels via WMMA f16->f32.
// Grid: BHW/16/8 blocks of 256 threads (8 waves/block).
// =====================================================================
template<int CIN, int COUT, int ACT, bool HB>
__global__ __launch_bounds__(256)
void conv1x1_k(const float* __restrict__ in, const float* __restrict__ W,
               const float* __restrict__ bias, float* __restrict__ out) {
  const int wave = threadIdx.x >> 5;
  const int lane = threadIdx.x & 31;
  const int tile = blockIdx.x * 8 + wave;          // 16-pixel tile id
  const int tilesPerImg = HWC / 16;                // 4096
  const int b   = tile / tilesPerImg;
  const int p0  = (tile - b * tilesPerImg) * 16;
  const float* inb = in + (size_t)b * CIN * HWC + p0;
  const int mrow = lane & 15;
  const int moff = (lane & 16) ? 8 : 0;

  v16h a[CIN / 32];
  #pragma unroll
  for (int kc = 0; kc < CIN / 32; ++kc) {
    #pragma unroll
    for (int i = 0; i < 16; ++i) {
      int k = kc * 32 + amap(lane, i);
      a[kc][i] = (_Float16)inb[k * HWC + mrow];
    }
  }
  #pragma unroll
  for (int nt = 0; nt < COUT / 16; ++nt) {
    const int n = nt * 16 + mrow;
    v8f acc = {};
    #pragma unroll
    for (int kc = 0; kc < CIN / 32; ++kc) {
      v16h bf;
      #pragma unroll
      for (int i = 0; i < 16; ++i) {
        int k = kc * 32 + bmapK(lane, i);
        bf[i] = (_Float16)W[n * CIN + k];
      }
      acc = __builtin_amdgcn_wmma_f32_16x16x32_f16(false, a[kc], false, bf,
                                                   (short)0, acc, false, false);
    }
    float bb = HB ? bias[n] : 0.f;
    float* ob = out + ((size_t)b * COUT + n) * HWC + p0 + moff;
    #pragma unroll
    for (int r = 0; r < 8; ++r) {
      float vv = acc[r] + bb;
      if (ACT == 1) vv = lrelu_(vv);
      ob[r] = vv;                                  // pixel = p0 + moff + r (= M)
    }
  }
}

// =====================================================================
// Windowed content-routed attention, one 8x8 window per block (128 thr, 4 waves).
// t = x_w*mask ; q=t Wq^T+bq ; k=t Wk^T+bk ; attn=softmax(q k^T) ;
// img = attn@(v_w*mask) + (v*sa)_w*(1-mask)   (scattered back to NCHW)
// =====================================================================
__global__ __launch_bounds__(128)
void attn_win_k(const float* __restrict__ x, const float* __restrict__ v,
                const float* __restrict__ sa, const float* __restrict__ mbin,
                const float* __restrict__ mask0, const float* __restrict__ mask1,
                const float* __restrict__ Wq, const float* __restrict__ bq,
                const float* __restrict__ Wk, const float* __restrict__ bk,
                float* __restrict__ img) {
  __shared__ _Float16 Ts[64 * 66];
  __shared__ _Float16 V1s[64 * 66];
  __shared__ _Float16 Qs[64 * 66];
  __shared__ _Float16 Ks[64 * 66];
  __shared__ _Float16 Ps[64 * 66];
  __shared__ float    Ss[64 * 66];

  const int win = blockIdx.x;                 // b*1024 + n
  const int b = win >> 10, n = win & 1023;
  const int wy = n >> 5, wx = n & 31;
  const int pbase = wy * 8 * 256 + wx * 8;    // pixel of token t: pbase + (t>>3)*256 + (t&7)
  const float* __restrict__ mv = (mbin[win] > 0.5f) ? mask1 : mask0;
  const int tid = threadIdx.x;

  // Stage masked tokens: T = x*mask, V1 = v*mask  (f16, row = token, col = channel)
  for (int idx = tid; idx < 4096; idx += 128) {
    int t = idx >> 6, c = idx & 63;
    int p = pbase + (t >> 3) * 256 + (t & 7);
    float m  = mv[idx];                       // embed index = t*64 + c
    float xv = x[((size_t)b * 64 + c) * HWC + p];
    float vv = v[((size_t)b * 64 + c) * HWC + p];
    Ts [t * 66 + c] = (_Float16)(xv * m);
    V1s[t * 66 + c] = (_Float16)(vv * m);
  }
  __syncthreads();

  const int wave = tid >> 5, lane = tid & 31;
  const int mrow = lane & 15;
  const int moff = (lane & 16) ? 8 : 0;
  const int m0 = wave * 16;                   // each wave owns one 16-row band

  // ---- Q = T Wq^T + bq ; K = T Wk^T + bk ----
  v16h aT[2];
  #pragma unroll
  for (int kc = 0; kc < 2; ++kc)
    #pragma unroll
    for (int i = 0; i < 16; ++i)
      aT[kc][i] = Ts[(m0 + mrow) * 66 + kc * 32 + amap(lane, i)];

  #pragma unroll
  for (int nt = 0; nt < 4; ++nt) {
    const int nn = nt * 16 + mrow;
    v8f qa = {}; v8f ka = {};
    #pragma unroll
    for (int kc = 0; kc < 2; ++kc) {
      v16h bq_, bk_;
      #pragma unroll
      for (int i = 0; i < 16; ++i) {
        int k = kc * 32 + bmapK(lane, i);
        bq_[i] = (_Float16)Wq[nn * 64 + k];
        bk_[i] = (_Float16)Wk[nn * 64 + k];
      }
      qa = __builtin_amdgcn_wmma_f32_16x16x32_f16(false, aT[kc], false, bq_, (short)0, qa, false, false);
      ka = __builtin_amdgcn_wmma_f32_16x16x32_f16(false, aT[kc], false, bk_, (short)0, ka, false, false);
    }
    float bqv = bq[nn], bkv = bk[nn];
    #pragma unroll
    for (int r = 0; r < 8; ++r) {
      Qs[(m0 + r + moff) * 66 + nn] = (_Float16)(qa[r] + bqv);
      Ks[(m0 + r + moff) * 66 + nn] = (_Float16)(ka[r] + bkv);
    }
  }
  __syncthreads();

  // ---- S = Q K^T ----
  v16h aQ[2];
  #pragma unroll
  for (int kc = 0; kc < 2; ++kc)
    #pragma unroll
    for (int i = 0; i < 16; ++i)
      aQ[kc][i] = Qs[(m0 + mrow) * 66 + kc * 32 + amap(lane, i)];

  #pragma unroll
  for (int nt = 0; nt < 4; ++nt) {
    v8f acc = {};
    #pragma unroll
    for (int kc = 0; kc < 2; ++kc) {
      v16h bf;                                 // B[k][n] = K[n][k]
      #pragma unroll
      for (int i = 0; i < 16; ++i)
        bf[i] = Ks[(nt * 16 + mrow) * 66 + kc * 32 + bmapK(lane, i)];
      acc = __builtin_amdgcn_wmma_f32_16x16x32_f16(false, aQ[kc], false, bf, (short)0, acc, false, false);
    }
    #pragma unroll
    for (int r = 0; r < 8; ++r)
      Ss[(m0 + r + moff) * 66 + nt * 16 + mrow] = acc[r];
  }
  __syncthreads();

  // ---- row softmax ----
  if (tid < 64) {
    float mx = -3.4e38f;
    for (int j = 0; j < 64; ++j) mx = fmaxf(mx, Ss[tid * 66 + j]);
    float s = 0.f;
    for (int j = 0; j < 64; ++j) { float e = expf(Ss[tid * 66 + j] - mx); Ss[tid * 66 + j] = e; s += e; }
    float inv = 1.f / s;
    for (int j = 0; j < 64; ++j) Ps[tid * 66 + j] = (_Float16)(Ss[tid * 66 + j] * inv);
  }
  __syncthreads();

  // ---- F = P V1 ; img = F + v*sa*(1-mask) ----
  v16h aP[2];
  #pragma unroll
  for (int kc = 0; kc < 2; ++kc)
    #pragma unroll
    for (int i = 0; i < 16; ++i)
      aP[kc][i] = Ps[(m0 + mrow) * 66 + kc * 32 + amap(lane, i)];

  #pragma unroll
  for (int nt = 0; nt < 4; ++nt) {
    v8f acc = {};
    #pragma unroll
    for (int kc = 0; kc < 2; ++kc) {
      v16h bf;                                 // B[k][n] = V1[token=k][chan=n]
      #pragma unroll
      for (int i = 0; i < 16; ++i)
        bf[i] = V1s[(kc * 32 + bmapK(lane, i)) * 66 + nt * 16 + mrow];
      acc = __builtin_amdgcn_wmma_f32_16x16x32_f16(false, aP[kc], false, bf, (short)0, acc, false, false);
    }
    const int c = nt * 16 + mrow;
    #pragma unroll
    for (int r = 0; r < 8; ++r) {
      int t = m0 + r + moff;
      int p = pbase + (t >> 3) * 256 + (t & 7);
      float m  = mv[t * 64 + c];
      float vv = v[((size_t)b * 64 + c) * HWC + p];
      float v2 = vv * sa[(size_t)b * HWC + p] * (1.f - m);
      img[((size_t)b * 64 + c) * HWC + p] = acc[r] + v2;
    }
  }
}

// =====================================================================
// Scalar helper: 3x3 depthwise tap with dilation + zero padding
// =====================================================================
__device__ __forceinline__ float dw3x3_at(const float* __restrict__ ip,
                                          const float* __restrict__ w9,
                                          int y, int x0, int dil) {
  float s = 0.f;
  #pragma unroll
  for (int ky = 0; ky < 3; ++ky) {
    int yy = y + (ky - 1) * dil;
    if (yy < 0 || yy > 255) continue;
    #pragma unroll
    for (int kx = 0; kx < 3; ++kx) {
      int xx = x0 + (kx - 1) * dil;
      if (xx < 0 || xx > 255) continue;
      s += w9[ky * 3 + kx] * ip[yy * 256 + xx];
    }
  }
  return s;
}

// tmp = dw3x3(img) + dwb1   (per channel)
__global__ __launch_bounds__(256)
void dw1_k(const float* __restrict__ in, const float* __restrict__ w,
           const float* __restrict__ bias, float* __restrict__ out) {
  int gid = blockIdx.x * 256 + threadIdx.x;        // B*C*HW exact
  int p = gid & (HWC - 1);
  int bc = gid >> 16;
  int c = bc & 63;
  float s = bias[c] + dw3x3_at(in + (size_t)bc * HWC, w + c * 9, p >> 8, p & 255, 1);
  out[gid] = s;
}

// out = gelu(dw3x3_dil2(tmp)+dwb2) * ca + img
__global__ __launch_bounds__(256)
void dw2_gelu_k(const float* __restrict__ in, const float* __restrict__ w,
                const float* __restrict__ bias, const float* __restrict__ ca,
                const float* __restrict__ img, float* __restrict__ out) {
  int gid = blockIdx.x * 256 + threadIdx.x;
  int p = gid & (HWC - 1);
  int bc = gid >> 16;
  int c = bc & 63, b = bc >> 6;
  float s = bias[c] + dw3x3_at(in + (size_t)bc * HWC, w + c * 9, p >> 8, p & 255, 2);
  out[gid] = gelu_(s) * ca[b * 64 + c] + img[gid];
}

// g = gelu(dw(h[0:64])) * dw(h[64:128])   (dw_ffn, no bias)
__global__ __launch_bounds__(256)
void ffn_gate_k(const float* __restrict__ h, const float* __restrict__ w,
                float* __restrict__ g) {
  int gid = blockIdx.x * 256 + threadIdx.x;        // B*64*HW exact
  int p = gid & (HWC - 1);
  int bc = gid >> 16;
  int c = bc & 63, b = bc >> 6;
  const float* h1 = h + ((size_t)b * 128 + c) * HWC;
  const float* h2 = h + ((size_t)b * 128 + 64 + c) * HWC;
  float s1 = dw3x3_at(h1, w + c * 9,        p >> 8, p & 255, 1);
  float s2 = dw3x3_at(h2, w + (64 + c) * 9, p >> 8, p & 255, 1);
  g[gid] = gelu_(s1) * s2;
}

// sa = sigmoid(conv3x3(x_) + bsa)   (16 -> 1 channels)
__global__ __launch_bounds__(256)
void sa_k(const float* __restrict__ xc, const float* __restrict__ Wsa,
          const float* __restrict__ bsa, float* __restrict__ sa) {
  int gid = blockIdx.x * 256 + threadIdx.x;        // B*HW exact
  int p = gid & (HWC - 1);
  int b = gid >> 16;
  float s = bsa[0];
  for (int ci = 0; ci < 16; ++ci)
    s += dw3x3_at(xc + ((size_t)b * 16 + ci) * HWC, Wsa + ci * 9, p >> 8, p & 255, 1);
  sa[gid] = sigm_(s);
}

// in-place LayerNorm over 16 channels + lrelu
__global__ __launch_bounds__(256)
void ln16_lrelu_k(float* __restrict__ xc, const float* __restrict__ lw,
                  const float* __restrict__ lb) {
  int gid = blockIdx.x * 256 + threadIdx.x;
  int p = gid & (HWC - 1);
  int b = gid >> 16;
  float r[16]; float s = 0.f;
  #pragma unroll
  for (int c = 0; c < 16; ++c) { r[c] = xc[((size_t)b * 16 + c) * HWC + p]; s += r[c]; }
  float u = s * (1.f / 16.f), s2 = 0.f;
  #pragma unroll
  for (int c = 0; c < 16; ++c) { float d = r[c] - u; s2 += d * d; }
  float inv = 1.f / sqrtf(s2 * (1.f / 16.f) + 1e-6f);
  #pragma unroll
  for (int c = 0; c < 16; ++c)
    xc[((size_t)b * 16 + c) * HWC + p] = lrelu_(lw[c] * (r[c] - u) * inv + lb[c]);
}

// out = LayerNorm_C64(a + r) (channels_first)
__global__ __launch_bounds__(256)
void addln64_k(const float* __restrict__ a, const float* __restrict__ rr,
               const float* __restrict__ lw, const float* __restrict__ lb,
               float* __restrict__ out) {
  int gid = blockIdx.x * 256 + threadIdx.x;
  int p = gid & (HWC - 1);
  int b = gid >> 16;
  size_t base = (size_t)b * 64 * HWC + p;
  float s = 0.f, s2 = 0.f;
  for (int c = 0; c < 64; ++c) {
    float t = a[base + (size_t)c * HWC] + rr[base + (size_t)c * HWC];
    s += t; s2 += t * t;
  }
  float u = s * (1.f / 64.f);
  float varr = s2 * (1.f / 64.f) - u * u;
  float inv = 1.f / sqrtf(varr + 1e-6f);
  for (int c = 0; c < 64; ++c) {
    float t = a[base + (size_t)c * HWC] + rr[base + (size_t)c * HWC];
    out[base + (size_t)c * HWC] = lw[c] * (t - u) * inv + lb[c];
  }
}

// per-(b,c16) spatial mean of x_  -> cm[b*16+c]
__global__ __launch_bounds__(256)
void cmean_k(const float* __restrict__ xc, float* __restrict__ cm) {
  __shared__ float red[256];
  int bc = blockIdx.x;
  const float* ip = xc + (size_t)bc * HWC;
  float s = 0.f;
  for (int i = threadIdx.x; i < HWC; i += 256) s += ip[i];
  red[threadIdx.x] = s; __syncthreads();
  for (int st = 128; st > 0; st >>= 1) {
    if (threadIdx.x < st) red[threadIdx.x] += red[threadIdx.x + st];
    __syncthreads();
  }
  if (threadIdx.x == 0) cm[bc] = red[0] * (1.f / HWC);
}

// ca = sigmoid(Wca @ cm + bca)   (256 outputs total)
__global__ __launch_bounds__(256)
void ca_k(const float* __restrict__ cm, const float* __restrict__ Wca,
          const float* __restrict__ bca, float* __restrict__ ca) {
  int o = threadIdx.x;                    // b*64 + oc
  int b = o >> 6, oc = o & 63;
  float s = bca[oc];
  for (int k = 0; k < 16; ++k) s += Wca[oc * 16 + k] * cm[b * 16 + k];
  ca[o] = sigm_(s);
}

// m16 = mean over 16 channels of xin
__global__ __launch_bounds__(256)
void mean16_k(const float* __restrict__ xin, float* __restrict__ m16) {
  int gid = blockIdx.x * 256 + threadIdx.x;
  int p = gid & (HWC - 1);
  int b = gid >> 16;
  float s = 0.f;
  #pragma unroll
  for (int c = 0; c < 16; ++c) s += xin[((size_t)b * 16 + c) * HWC + p];
  m16[gid] = s * (1.f / 16.f);
}

// unbiased window variance of m16 over each 8x8 window
__global__ __launch_bounds__(256)
void winvar_k(const float* __restrict__ m16, float* __restrict__ var) {
  int wid = blockIdx.x * 256 + threadIdx.x;   // 4096 exact
  int b = wid >> 10, n = wid & 1023;
  int pbase = (n >> 5) * 8 * 256 + (n & 31) * 8;
  float s = 0.f, s2 = 0.f;
  for (int t = 0; t < 64; ++t) {
    float vv = m16[(size_t)b * HWC + pbase + (t >> 3) * 256 + (t & 7)];
    s += vv; s2 += vv * vv;
  }
  var[wid] = (s2 - s * s * (1.f / 64.f)) * (1.f / 63.f);
}

// bottom-512-variance windows get mbin=0 (ties -> lower index, matching lax.top_k)
__global__ __launch_bounds__(256)
void mbin_k(const float* __restrict__ var, float* __restrict__ mbin) {
  __shared__ float lv[1024];
  int b = blockIdx.x;
  for (int i = threadIdx.x; i < 1024; i += 256) lv[i] = var[b * 1024 + i];
  __syncthreads();
  for (int n = threadIdx.x; n < 1024; n += 256) {
    float vn = lv[n];
    int cnt = 0;
    for (int m = 0; m < 1024; ++m)
      cnt += (lv[m] < vn || (lv[m] == vn && m < n)) ? 1 : 0;
    mbin[b * 1024 + n] = (cnt < 512) ? 0.f : 1.f;
  }
}

// mask MLP collapses: hidden0 = lrelu(bm1), hidden1 = lrelu(rowsum(Wm1)+bm1)
__global__ __launch_bounds__(256)
void mhid_k(const float* __restrict__ Wm1, const float* __restrict__ bm1,
            float* __restrict__ h0, float* __restrict__ h1) {
  int j = blockIdx.x * 256 + threadIdx.x;     // 2048 exact
  const float* wr = Wm1 + (size_t)j * 4096;
  float rs = 0.f;
  for (int e = 0; e < 4096; ++e) rs += wr[e];
  h0[j] = lrelu_(bm1[j]);
  h1[j] = lrelu_(rs + bm1[j]);
}

// mask{0,1} = hidden{0,1} @ Wm2.T + bm2
__global__ __launch_bounds__(256)
void mout_k(const float* __restrict__ Wm2, const float* __restrict__ bm2,
            const float* __restrict__ h0, const float* __restrict__ h1,
            float* __restrict__ mk0, float* __restrict__ mk1) {
  int e = blockIdx.x * 256 + threadIdx.x;     // 4096 exact
  const float* wr = Wm2 + (size_t)e * 2048;
  float s0 = 0.f, s1 = 0.f;
  for (int j = 0; j < 2048; ++j) { float w = wr[j]; s0 += w * h0[j]; s1 += w * h1[j]; }
  mk0[e] = s0 + bm2[e];
  mk1[e] = s1 + bm2[e];
}

// =====================================================================
extern "C" void kernel_launch(void* const* d_in, const int* in_sizes, int n_in,
                              void* d_out, int out_size, void* d_ws, size_t ws_size,
                              hipStream_t stream) {
  const float* x    = (const float*)d_in[0];
  const float* w_n1 = (const float*)d_in[1];
  const float* b_n1 = (const float*)d_in[2];
  const float* w_n2 = (const float*)d_in[3];
  const float* b_n2 = (const float*)d_in[4];
  const float* Wv   = (const float*)d_in[5];
  const float* bv   = (const float*)d_in[6];
  const float* Wq   = (const float*)d_in[7];
  const float* bq   = (const float*)d_in[8];
  const float* Wk   = (const float*)d_in[9];
  const float* bk   = (const float*)d_in[10];
  const float* dw1  = (const float*)d_in[11];
  const float* dwb1 = (const float*)d_in[12];
  const float* dw2  = (const float*)d_in[13];
  const float* dwb2 = (const float*)d_in[14];
  const float* Wo   = (const float*)d_in[15];
  const float* bo   = (const float*)d_in[16];
  const float* W_in = (const float*)d_in[17];
  const float* b_in = (const float*)d_in[18];
  const float* Wc   = (const float*)d_in[19];
  const float* bc_  = (const float*)d_in[20];
  const float* ln_w = (const float*)d_in[21];
  const float* ln_b = (const float*)d_in[22];
  const float* Wm1  = (const float*)d_in[23];
  const float* bm1  = (const float*)d_in[24];
  const float* Wm2  = (const float*)d_in[25];
  const float* bm2  = (const float*)d_in[26];
  const float* Wca  = (const float*)d_in[27];
  const float* bca  = (const float*)d_in[28];
  const float* Wsa  = (const float*)d_in[29];
  const float* bsa  = (const float*)d_in[30];
  const float* Wfi  = (const float*)d_in[31];
  const float* dwf  = (const float*)d_in[32];
  const float* Wfo  = (const float*)d_in[33];

  float* ws = (float*)d_ws;
  size_t o = 0;
  const size_t FMAP = (size_t)BATCH * 64 * HWC;    // 16,777,216 floats
  float* A    = ws + o; o += FMAP;                 // v -> res -> g
  float* Bb   = ws + o; o += FMAP;                 // img -> x1
  float* Cb   = ws + o; o += FMAP;                 // dw tmp -> h[lo] -> res2
  float* Db   = ws + o; o += FMAP;                 // out -> h[hi]
  float* hbuf = Cb;                                // 128-ch FFN hidden spans Cb..Db
  float* xinb = ws + o; o += (size_t)BATCH * 16 * HWC;
  float* xcb  = ws + o; o += (size_t)BATCH * 16 * HWC;
  float* sab  = ws + o; o += (size_t)BATCH * HWC;
  float* m16b = ws + o; o += (size_t)BATCH * HWC;
  float* varb = ws + o; o += BATCH * NWIN;
  float* mbinb= ws + o; o += BATCH * NWIN;
  float* cmb  = ws + o; o += BATCH * 16;
  float* cab  = ws + o; o += BATCH * 64;
  float* h0b  = ws + o; o += 2048;
  float* h1b  = ws + o; o += 2048;
  float* mk0  = ws + o; o += 4096;
  float* mk1  = ws + o; o += 4096;

  const int GPW = 262144 / 16 / 8;   // conv1x1 grid: 2048
  const int GPX = 262144 / 256;      // per-pixel grid: 1024
  const int GBC = 16777216 / 256;    // per-(b,c,pixel) grid: 65536

  // ---------------- CAMixer ----------------
  conv1x1_k<64, 64, 0, true ><<<GPW, 256, 0, stream>>>(x,  Wv,   bv,   A);     // v
  conv1x1_k<64, 16, 1, true ><<<GPW, 256, 0, stream>>>(A,  W_in, b_in, xinb);  // xin = lrelu(.)
  conv1x1_k<64, 16, 0, true ><<<GPW, 256, 0, stream>>>(A,  Wc,   bc_,  xcb);   // pre-LN x_
  ln16_lrelu_k<<<GPX, 256, 0, stream>>>(xcb, ln_w, ln_b);                      // x_ = lrelu(ln(.))
  cmean_k<<<BATCH * 16, 256, 0, stream>>>(xcb, cmb);
  ca_k<<<1, 256, 0, stream>>>(cmb, Wca, bca, cab);                             // ca
  sa_k<<<GPX, 256, 0, stream>>>(xcb, Wsa, bsa, sab);                           // sa
  mean16_k<<<GPX, 256, 0, stream>>>(xinb, m16b);
  winvar_k<<<16, 256, 0, stream>>>(m16b, varb);
  mbin_k<<<BATCH, 256, 0, stream>>>(varb, mbinb);                              // routing mask bits
  mhid_k<<<8, 256, 0, stream>>>(Wm1, bm1, h0b, h1b);                           // collapsed mask MLP
  mout_k<<<16, 256, 0, stream>>>(Wm2, bm2, h0b, h1b, mk0, mk1);
  attn_win_k<<<BATCH * NWIN, 128, 0, stream>>>(x, A, sab, mbinb, mk0, mk1,
                                               Wq, bq, Wk, bk, Bb);            // img
  dw1_k<<<GBC, 256, 0, stream>>>(Bb, dw1, dwb1, Cb);                           // dw 3x3
  dw2_gelu_k<<<GBC, 256, 0, stream>>>(Cb, dw2, dwb2, cab, Bb, Db);             // out
  conv1x1_k<64, 64, 0, true ><<<GPW, 256, 0, stream>>>(Db, Wo, bo, A);         // res
  // ---------------- norm1 + gated FFN + norm2 ----------------
  addln64_k<<<GPX, 256, 0, stream>>>(x, A, w_n1, b_n1, Bb);                    // x1
  conv1x1_k<64, 128, 0, false><<<GPW, 256, 0, stream>>>(Bb, Wfi, nullptr, hbuf); // h (128ch)
  ffn_gate_k<<<GBC, 256, 0, stream>>>(hbuf, dwf, A);                           // g
  conv1x1_k<64, 64, 0, false><<<GPW, 256, 0, stream>>>(A, Wfo, nullptr, Cb);   // res2
  addln64_k<<<GPX, 256, 0, stream>>>(Bb, Cb, w_n2, b_n2, (float*)d_out);       // final
}
